// MHA_71150428226240
// MI455X (gfx1250) — matrix operand order repaired and checked by reading, not compiled, not run on hardware
//
#include <hip/hip_runtime.h>
#include <hip/hip_bf16.h>
#include <math.h>

typedef _Float16 half_t;
typedef __attribute__((ext_vector_type(16))) _Float16 v16h;
typedef __attribute__((ext_vector_type(8)))  float    v8f;
typedef __attribute__((ext_vector_type(4)))  int      v4i;

#define BATCH  4
#define SEQ    2048
#define DMODEL 1024
#define NHEADS 16
#define HD     64
#define ROWS   (BATCH * SEQ)   // 8192

__device__ __forceinline__ v8f v8f_zero() {
  v8f r;
#pragma unroll
  for (int i = 0; i < 8; ++i) r[i] = 0.0f;
  return r;
}

__device__ __forceinline__ v8f wmma_f16(v16h a, v16h b, v8f c) {
  // D = A(16x32 f16) * B(32x16 f16) + C(16x16 f32)
  return __builtin_amdgcn_wmma_f32_16x16x32_f16(false, a, false, b, (short)0, c,
                                                false, false);
}

// Async 16B global -> LDS copy (GLOBAL_LOAD_ASYNC_TO_LDS_B128, ASYNCcnt).
__device__ __forceinline__ void async_copy_b128(const half_t* g, half_t* l) {
  __builtin_amdgcn_global_load_async_to_lds_b128((v4i*)g, (v4i*)l, 0, 0);
}

__device__ __forceinline__ void wait_async0() {
  __builtin_amdgcn_s_wait_asynccnt(0);
}

// ---------------------------------------------------------------------------
// Elementwise fp32 -> f16 convert (for x -> xh).
// ---------------------------------------------------------------------------
__global__ __launch_bounds__(256)
void convert_f16(const float* __restrict__ in, half_t* __restrict__ out,
                 int nvec4) {
  const int i = blockIdx.x * blockDim.x + threadIdx.x;
  if (i < nvec4) {
    const float4 v = ((const float4*)in)[i];
    half_t* o = out + (size_t)i * 4;
    o[0] = (half_t)v.x; o[1] = (half_t)v.y;
    o[2] = (half_t)v.z; o[3] = (half_t)v.w;
  }
}

// ---------------------------------------------------------------------------
// Tiled transpose + fp32 -> f16: outT[c][k] = (f16) in[k][c], 1024x1024.
// ---------------------------------------------------------------------------
__global__ __launch_bounds__(256)
void transpose_f16(const float* __restrict__ in, half_t* __restrict__ outT) {
  __shared__ float tile[32][33];
  const int bx = blockIdx.x * 32;  // c base
  const int by = blockIdx.y * 32;  // k base
  const int tx = threadIdx.x;      // 0..31
  const int ty = threadIdx.y;      // 0..7
#pragma unroll
  for (int i = 0; i < 32; i += 8)
    tile[ty + i][tx] = in[(size_t)(by + ty + i) * DMODEL + bx + tx];
  __syncthreads();
#pragma unroll
  for (int i = 0; i < 32; i += 8)
    outT[(size_t)(bx + ty + i) * DMODEL + by + tx] = (half_t)tile[tx][ty + i];
}

// ---------------------------------------------------------------------------
// Tiled GEMM: Y[8192 x 1024] = Xh[8192 x 1024] @ WT[1024 x 1024]^T
// (WT stored col-major: WT[c][k], both operands f16).
// Block: 256 threads (8 waves), 128x128 tile, K stepped by 32,
// double-buffered LDS filled with async global->LDS b128 copies.
// mode 0: out f16, head-split   [b*H+h][n][64]        (Q, K)
// mode 1: out f16, head-split T [b*H+h][64][n]        (V^T)
// mode 2: out f32 row-major + bias                    (final projection)
// ---------------------------------------------------------------------------
__global__ __launch_bounds__(256)
void gemm1024_wmma(const half_t* __restrict__ Xh, const half_t* __restrict__ WT,
                   const float* __restrict__ bias, half_t* __restrict__ outH,
                   float* __restrict__ outF, int mode) {
  constexpr int LDT = 48;  // 32 + 16 halves pad: 96B row stride, 16B aligned
  __shared__ __align__(16) half_t As[2][128][LDT];
  __shared__ __align__(16) half_t Bs[2][128][LDT];  // Bs[col][k]

  const int tid  = threadIdx.x;
  const int wid  = tid >> 5;
  const int lane = tid & 31;
  const int l16  = lane & 15;
  const int lh   = lane >> 4;

  const int colBase = blockIdx.x * 128;
  const int rowBase = blockIdx.y * 128;
  const int wm = (wid & 3) * 32;   // wave row offset in block tile
  const int wn = (wid >> 2) * 64;  // wave col offset in block tile

  v8f acc[2][4];
#pragma unroll
  for (int mt = 0; mt < 2; ++mt)
#pragma unroll
    for (int nt = 0; nt < 4; ++nt) acc[mt][nt] = v8f_zero();

  // Each thread issues 2x2 async b128 copies per stage (8KB + 8KB per buffer).
  auto stage = [&](int kt, int buf) {
#pragma unroll
    for (int i = 0; i < 2; ++i) {
      const int c   = tid + 256 * i;   // 0..511 chunk id
      const int row = c >> 2;          // 0..127
      const int off = (c & 3) * 8;     // half offset within 32-half row
      async_copy_b128(Xh + (size_t)(rowBase + row) * DMODEL + kt * 32 + off,
                      &As[buf][row][off]);
      async_copy_b128(WT + (size_t)(colBase + row) * DMODEL + kt * 32 + off,
                      &Bs[buf][row][off]);
    }
  };

  stage(0, 0);
  wait_async0();
  __syncthreads();

  for (int kt = 0; kt < 32; ++kt) {
    const int cur = kt & 1;
    if (kt + 1 < 32) stage(kt + 1, cur ^ 1);

    // A fragments (16x32): lane row = l16; K striped per ISA 7.12.2
    v16h a[2], b[4];
#pragma unroll
    for (int mt = 0; mt < 2; ++mt) {
      const half_t* ar = &As[cur][wm + mt * 16 + l16][0];
#pragma unroll
      for (int e = 0; e < 8; ++e) {
        a[mt][e]     = ar[8 * lh + e];
        a[mt][e + 8] = ar[16 + 8 * lh + e];
      }
    }
    // B fragments (32x16): lane col = l16; lanes 0-15 K=0..7,16..23; 16-31 K=8..15,24..31
#pragma unroll
    for (int nt = 0; nt < 4; ++nt) {
      const half_t* br = &Bs[cur][wn + nt * 16 + l16][0];
#pragma unroll
      for (int e = 0; e < 8; ++e) {
        b[nt][e]     = br[8 * lh + e];
        b[nt][e + 8] = br[16 + 8 * lh + e];
      }
    }
#pragma unroll
    for (int mt = 0; mt < 2; ++mt)
#pragma unroll
      for (int nt = 0; nt < 4; ++nt)
        acc[mt][nt] = wmma_f16(a[mt], b[nt], acc[mt][nt]);

    wait_async0();   // staged tile for kt+1 fully in LDS
    __syncthreads();
  }

  // Epilogue: C layout = lane col l16, row = j + 8*lh
#pragma unroll
  for (int mt = 0; mt < 2; ++mt)
#pragma unroll
    for (int nt = 0; nt < 4; ++nt)
#pragma unroll
      for (int j = 0; j < 8; ++j) {
        const int gr = rowBase + wm + mt * 16 + j + 8 * lh;
        const int gc = colBase + wn + nt * 16 + l16;
        const float v = acc[mt][nt][j];
        if (mode == 2) {
          outF[(size_t)gr * DMODEL + gc] = v + bias[gc];
        } else {
          const int bb = gr >> 11, n = gr & (SEQ - 1);
          const int hh = gc >> 6,  d = gc & (HD - 1);
          const size_t bh = (size_t)bb * NHEADS + hh;
          if (mode == 0) outH[(bh * SEQ + n) * HD + d] = (half_t)v;
          else           outH[(bh * HD + d) * SEQ + n] = (half_t)v;
        }
      }
}

// ---------------------------------------------------------------------------
// Flash attention (causal), one wave = 16 query rows, keys in blocks of 64.
// Q,K in [bh][n][64] f16, V transposed [bh][64][n] f16; ctx out f16
// in [b][n][1024] (heads re-interleaved).
// ---------------------------------------------------------------------------
__global__ __launch_bounds__(256)
void flash_attn_wmma(const half_t* __restrict__ Q, const half_t* __restrict__ Km,
                     const half_t* __restrict__ Vt, half_t* __restrict__ ctx) {
  constexpr int LDP = 72;  // 64 + 8 halves pad, 144B stride (16B aligned)
  __shared__ __align__(16) half_t Pl[8][16][LDP];

  const int tid  = threadIdx.x;
  const int wid  = tid >> 5;
  const int lane = tid & 31;
  const int l16  = lane & 15;
  const int lh   = lane >> 4;

  const int bh = blockIdx.y;                  // 0..63
  const int q0 = blockIdx.x * 128 + wid * 16; // query block base for this wave

  const half_t* Qb = Q  + (size_t)bh * SEQ * HD;
  const half_t* Kb = Km + (size_t)bh * SEQ * HD;
  const half_t* Vb = Vt + (size_t)bh * HD * SEQ;

  // Q A-fragments: qa[0] = d 0..31, qa[1] = d 32..63
  v16h qa[2];
  {
    const half_t* qr = Qb + (size_t)(q0 + l16) * HD;
#pragma unroll
    for (int e = 0; e < 8; ++e) {
      qa[0][e]     = qr[8 * lh + e];
      qa[0][e + 8] = qr[16 + 8 * lh + e];
      qa[1][e]     = qr[32 + 8 * lh + e];
      qa[1][e + 8] = qr[48 + 8 * lh + e];
    }
  }

  v8f O[4];
#pragma unroll
  for (int nt = 0; nt < 4; ++nt) O[nt] = v8f_zero();
  float mrow[8], lrow[8];
#pragma unroll
  for (int j = 0; j < 8; ++j) { mrow[j] = -__builtin_inff(); lrow[j] = 0.0f; }

  const float scale = 0.125f;  // 1/sqrt(64)

  for (int kb = 0; kb < q0 + 16; kb += 64) {
    // ---- scores: S(16x64) = Q(16x64) @ K^T(64x64-block) ----
    v8f S[4];
#pragma unroll
    for (int t = 0; t < 4; ++t) {
      v16h kf0, kf1;  // B frags: lane = key col, elements = d
      const half_t* kr = Kb + (size_t)(kb + t * 16 + l16) * HD;
#pragma unroll
      for (int e = 0; e < 8; ++e) {
        kf0[e]     = kr[8 * lh + e];       // d 0..31
        kf0[e + 8] = kr[16 + 8 * lh + e];
        kf1[e]     = kr[32 + 8 * lh + e];  // d 32..63
        kf1[e + 8] = kr[48 + 8 * lh + e];
      }
      v8f s = v8f_zero();
      s = wmma_f16(qa[0], kf0, s);
      s = wmma_f16(qa[1], kf1, s);
      S[t] = s;
    }

    // ---- scale + causal mask + online softmax ----
    float mnew[8];
#pragma unroll
    for (int j = 0; j < 8; ++j) mnew[j] = mrow[j];
#pragma unroll
    for (int t = 0; t < 4; ++t)
#pragma unroll
      for (int j = 0; j < 8; ++j) {
        const int col = kb + t * 16 + l16;
        const int row = q0 + j + 8 * lh;
        float v = S[t][j] * scale;
        if (col > row) v = -__builtin_inff();
        S[t][j] = v;
        mnew[j] = fmaxf(mnew[j], v);
      }
#pragma unroll
    for (int j = 0; j < 8; ++j)
#pragma unroll
      for (int m = 1; m <= 8; m <<= 1)
        mnew[j] = fmaxf(mnew[j], __shfl_xor(mnew[j], m, 32));

    float fac[8], rsum[8];
#pragma unroll
    for (int j = 0; j < 8; ++j) {
      fac[j] = __expf(mrow[j] - mnew[j]);
      rsum[j] = 0.0f;
      mrow[j] = mnew[j];
    }
#pragma unroll
    for (int t = 0; t < 4; ++t)
#pragma unroll
      for (int j = 0; j < 8; ++j) {
        const float p = __expf(S[t][j] - mnew[j]);
        S[t][j] = p;
        rsum[j] += p;
      }
#pragma unroll
    for (int j = 0; j < 8; ++j)
#pragma unroll
      for (int m = 1; m <= 8; m <<= 1)
        rsum[j] += __shfl_xor(rsum[j], m, 32);
#pragma unroll
    for (int j = 0; j < 8; ++j) lrow[j] = lrow[j] * fac[j] + rsum[j];
#pragma unroll
    for (int nt = 0; nt < 4; ++nt)
#pragma unroll
      for (int j = 0; j < 8; ++j) O[nt][j] *= fac[j];

    // ---- P: C-layout -> A-layout via per-wave LDS patch ----
#pragma unroll
    for (int t = 0; t < 4; ++t)
#pragma unroll
      for (int j = 0; j < 8; ++j)
        Pl[wid][j + 8 * lh][t * 16 + l16] = (half_t)S[t][j];

    v16h pa[2];
    {
      const half_t* pr = &Pl[wid][l16][0];
#pragma unroll
      for (int e = 0; e < 8; ++e) {
        pa[0][e]     = pr[8 * lh + e];
        pa[0][e + 8] = pr[16 + 8 * lh + e];
        pa[1][e]     = pr[32 + 8 * lh + e];
        pa[1][e + 8] = pr[48 + 8 * lh + e];
      }
    }

    // ---- O += P(16x64) @ V(64x64-block) ----
#pragma unroll
    for (int nt = 0; nt < 4; ++nt) {
      v16h vb0, vb1;  // B frags from V^T: lane = d col, elements = key
      const half_t* vr = Vb + (size_t)(nt * 16 + l16) * SEQ + kb;
#pragma unroll
      for (int e = 0; e < 8; ++e) {
        vb0[e]     = vr[8 * lh + e];       // keys kb+0..31
        vb0[e + 8] = vr[16 + 8 * lh + e];
        vb1[e]     = vr[32 + 8 * lh + e];  // keys kb+32..63
        vb1[e + 8] = vr[48 + 8 * lh + e];
      }
      O[nt] = wmma_f16(pa[0], vb0, O[nt]);
      O[nt] = wmma_f16(pa[1], vb1, O[nt]);
    }
  }

  // ---- normalize + write ctx [b][n][H*64] ----
  const int bb = bh >> 4, hh = bh & 15;
#pragma unroll
  for (int nt = 0; nt < 4; ++nt)
#pragma unroll
    for (int j = 0; j < 8; ++j) {
      const int row = q0 + j + 8 * lh;
      const float ov = O[nt][j] / lrow[j];
      ctx[((size_t)(bb * SEQ + row)) * DMODEL + hh * HD + nt * 16 + l16] =
          (half_t)ov;
    }
}

// ---------------------------------------------------------------------------
extern "C" void kernel_launch(void* const* d_in, const int* in_sizes, int n_in,
                              void* d_out, int out_size, void* d_ws,
                              size_t ws_size, hipStream_t stream) {
  (void)in_sizes; (void)n_in; (void)out_size; (void)ws_size;

  const float* x  = (const float*)d_in[0];
  const float* Wq = (const float*)d_in[1];
  const float* Wk = (const float*)d_in[2];
  const float* Wv = (const float*)d_in[3];
  const float* Wo = (const float*)d_in[4];
  const float* bo = (const float*)d_in[5];
  float* out = (float*)d_out;

  // Workspace (f16): xh | WqT WkT WvT WoT | Q | K | V^T | ctx  = 88 MB
  const size_t perT = (size_t)BATCH * NHEADS * SEQ * HD;  // 8,388,608 halves
  const size_t perW = (size_t)DMODEL * DMODEL;            // 1,048,576 halves
  half_t* xh  = (half_t*)d_ws;
  half_t* WqT = xh  + perT;
  half_t* WkT = WqT + perW;
  half_t* WvT = WkT + perW;
  half_t* WoT = WvT + perW;
  half_t* Q   = WoT + perW;
  half_t* Kh  = Q   + perT;
  half_t* Vt  = Kh  + perT;
  half_t* ctx = Vt  + perT;

  // --- precision staging: x -> f16, W* -> f16 transposed ---
  {
    const int nvec4 = (int)((size_t)ROWS * DMODEL / 4);
    convert_f16<<<dim3((nvec4 + 255) / 256), dim3(256), 0, stream>>>(x, xh,
                                                                     nvec4);
    dim3 tb(32, 8), tg(DMODEL / 32, DMODEL / 32);
    transpose_f16<<<tg, tb, 0, stream>>>(Wq, WqT);
    transpose_f16<<<tg, tb, 0, stream>>>(Wk, WkT);
    transpose_f16<<<tg, tb, 0, stream>>>(Wv, WvT);
    transpose_f16<<<tg, tb, 0, stream>>>(Wo, WoT);
  }

  dim3 blk(256);
  dim3 gridG(DMODEL / 128, ROWS / 128);  // (8, 64)

  gemm1024_wmma<<<gridG, blk, 0, stream>>>(xh, WqT, (const float*)nullptr, Q,
                                           (float*)nullptr, 0);
  gemm1024_wmma<<<gridG, blk, 0, stream>>>(xh, WkT, (const float*)nullptr, Kh,
                                           (float*)nullptr, 0);
  gemm1024_wmma<<<gridG, blk, 0, stream>>>(xh, WvT, (const float*)nullptr, Vt,
                                           (float*)nullptr, 1);

  dim3 gridF(SEQ / 128, BATCH * NHEADS);  // (16, 64)
  flash_attn_wmma<<<gridF, blk, 0, stream>>>(Q, Kh, Vt, ctx);

  gemm1024_wmma<<<gridG, blk, 0, stream>>>(ctx, WoT, bo, (half_t*)nullptr, out,
                                           2);
}
